// STSA_36679020708366
// MI455X (gfx1250) — matrix operand order repaired and checked by reading, not compiled, not run on hardware
//
#include <hip/hip_runtime.h>
#include <stdint.h>
#include <stddef.h>

// ---------------- problem constants ----------------
#define B_  8
#define S_  512
#define E_  512
#define H_  8
#define D_  64
#define F_  2048
#define P_  64
#define RC_ 129      // 2P+1 distinct relative-position rows
#define RP_ 160      // RC padded to a multiple of 32 (WMMA K-step)
#define BS_ (B_*S_)  // 4096 token rows
#define BH_ (B_*H_)  // 64 (batch,head) pairs

typedef __attribute__((ext_vector_type(16))) __bf16 v16bf;
typedef __attribute__((ext_vector_type(8)))  float  v8f;

union FragU { v16bf v; uint32_t u[8]; };

// A-matrix 16x32 bf16 fragment (row-major A, lda in elements).
// ISA layout: lane m=lane&15, laneGrp g=lane>>4; element e=2v+p ->
// k = 16*(v/4) + 8*g + 2*(v%4) + p
__device__ inline v16bf load_a_frag(const __bf16* A, int lda, int lane, int k0) {
  int m = lane & 15;
  int g = (lane >> 4) & 1;
  const __bf16* base = A + (size_t)m * lda + k0 + g * 8;
  FragU t;
#pragma unroll
  for (int v = 0; v < 8; ++v) {
    int koff = ((v >> 2) << 4) + ((v & 3) << 1);
    t.u[v] = *(const uint32_t*)(base + koff);
  }
  return t.v;
}

// B-matrix 32x16 bf16 fragment from an n-major (transposed) operand:
// Bt[n*ldb + k].  ISA layout: n=lane&15, g=lane>>4; k = 16*(v/4) + 8*g + 2*(v%4) + p
__device__ inline v16bf load_b_frag(const __bf16* Bt, int ldb, int lane, int k0) {
  int n = lane & 15;
  int g = (lane >> 4) & 1;
  const __bf16* base = Bt + (size_t)n * ldb + k0 + g * 8;
  FragU t;
#pragma unroll
  for (int v = 0; v < 8; ++v) t.u[v] = *(const uint32_t*)(base + ((v >> 2) << 4) + ((v & 3) << 1));
  return t.v;
}

__device__ inline v8f wmma_bf16(v16bf a, v16bf b, v8f c) {
  return __builtin_amdgcn_wmma_f32_16x16x32_bf16(false, a, false, b, (short)0, c,
                                                 false, false);
}

// ---------------- conversion kernels ----------------
__global__ __launch_bounds__(256) void stsa_cvt_bf16(const float* src, __bf16* dst, int n) {
  int i = blockIdx.x * 256 + threadIdx.x;
  if (i < n) dst[i] = (__bf16)src[i];
}

// src is K x N (row-major), dst is N x K (n-major bf16)
__global__ __launch_bounds__(256) void stsa_transpose_bf16(const float* src, __bf16* dst,
                                                           int K, int N) {
  int i = blockIdx.x * 256 + threadIdx.x;
  if (i >= K * N) return;
  int n = i / K, k = i - n * K;
  dst[i] = (__bf16)src[(size_t)k * N + n];
}

// pos_emb (129 x 64) -> posB (RP x 64, r-major, zero padded) and
// posT (64 x RP, d-major, zero padded)
__global__ __launch_bounds__(256) void stsa_pos_cvt(const float* pos, __bf16* posB,
                                                    __bf16* posT) {
  int i = blockIdx.x * 256 + threadIdx.x;
  if (i >= RP_ * D_) return;
  int r = i / D_, d = i - r * D_;
  float v = (r < RC_) ? pos[r * D_ + d] : 0.f;
  posB[r * D_ + d] = (__bf16)v;
  posT[d * RP_ + r] = (__bf16)v;
}

// ---------------- QKV projection (32x32 tile per wave) ----------------
// out (vmode=0): head-major  q[((b*H+h)*S + s)*D + d]
// out (vmode=1): transposed  v[((b*H+h)*D + d)*S + s]
__global__ __launch_bounds__(128) void stsa_gemm_qkv(const __bf16* xb, const __bf16* wT,
                                                     const float* bias, __bf16* out,
                                                     int vmode) {
  int m0 = blockIdx.x * 32;
  int wave = threadIdx.x >> 5;
  int lane = threadIdx.x & 31;
  int n0 = blockIdx.y * 128 + wave * 32;
  const __bf16* A0 = xb + (size_t)m0 * E_;
  const __bf16* A1 = A0 + (size_t)16 * E_;
  const __bf16* B0 = wT + (size_t)n0 * E_;
  const __bf16* B1 = B0 + (size_t)16 * E_;
  v8f acc00 = {}, acc01 = {}, acc10 = {}, acc11 = {};
  for (int k0 = 0; k0 < E_; k0 += 32) {
    v16bf a0 = load_a_frag(A0, E_, lane, k0);
    v16bf a1 = load_a_frag(A1, E_, lane, k0);
    v16bf b0 = load_b_frag(B0, E_, lane, k0);
    v16bf b1 = load_b_frag(B1, E_, lane, k0);
    acc00 = wmma_bf16(a0, b0, acc00);
    acc01 = wmma_bf16(a0, b1, acc01);
    acc10 = wmma_bf16(a1, b0, acc10);
    acc11 = wmma_bf16(a1, b1, acc11);
  }
  int g = (lane >> 4) & 1;
  int nl = lane & 15;
#pragma unroll
  for (int ni = 0; ni < 2; ++ni) {
    int n = n0 + ni * 16 + nl;
    int h = n >> 6, d = n & 63;
    float bv = bias[n];
#pragma unroll
    for (int mi = 0; mi < 2; ++mi) {
      v8f acc = (mi == 0) ? (ni == 0 ? acc00 : acc01) : (ni == 0 ? acc10 : acc11);
#pragma unroll
      for (int r = 0; r < 8; ++r) {
        int m = m0 + mi * 16 + r + 8 * g;
        int b0i = m >> 9, s = m & 511;
        int bh = b0i * H_ + h;
        float v = acc[r] + bv;
        if (vmode == 0) out[((size_t)bh * S_ + s) * D_ + d] = (__bf16)v;
        else            out[((size_t)bh * D_ + d) * S_ + s] = (__bf16)v;
      }
    }
  }
}

// ---------------- generic GEMM, 32x32 register tile per wave ----------------
// A row-major bf16 (lda), Bt n-major bf16 (ldb).  N must be a multiple of 32.
// EPI 0: outF = acc                  (q . pos^T)
// EPI 1: outF = acc + bias + res     (O-proj, FFN2)
// EPI 2: outB = bf16(relu(acc+bias)) (FFN1)
template <int EPI>
__global__ __launch_bounds__(128) void stsa_gemm(const __bf16* A, const __bf16* Bt,
                                                 const float* bias, const float* res,
                                                 float* outF, __bf16* outB,
                                                 int N, int K, int lda, int ldb) {
  int m0 = blockIdx.x * 32;
  int wave = threadIdx.x >> 5;
  int lane = threadIdx.x & 31;
  int n0 = blockIdx.y * 128 + wave * 32;
  if (n0 >= N) return;  // wave-uniform
  const __bf16* A0 = A + (size_t)m0 * lda;
  const __bf16* A1 = A0 + (size_t)16 * lda;
  const __bf16* B0 = Bt + (size_t)n0 * ldb;
  const __bf16* B1 = B0 + (size_t)16 * ldb;
  int ml = lane & 15;
  v8f acc00 = {}, acc01 = {}, acc10 = {}, acc11 = {};
  for (int k0 = 0; k0 < K; k0 += 32) {
    if (k0 + 128 < K) {  // stream-ahead prefetch (global_prefetch_b8)
      __builtin_prefetch(A0 + (size_t)ml * lda + k0 + 128, 0, 3);
      __builtin_prefetch(A1 + (size_t)ml * lda + k0 + 128, 0, 3);
    }
    v16bf a0 = load_a_frag(A0, lda, lane, k0);
    v16bf a1 = load_a_frag(A1, lda, lane, k0);
    v16bf b0 = load_b_frag(B0, ldb, lane, k0);
    v16bf b1 = load_b_frag(B1, ldb, lane, k0);
    acc00 = wmma_bf16(a0, b0, acc00);
    acc01 = wmma_bf16(a0, b1, acc01);
    acc10 = wmma_bf16(a1, b0, acc10);
    acc11 = wmma_bf16(a1, b1, acc11);
  }
  int g = (lane >> 4) & 1;
  int nl = lane & 15;
#pragma unroll
  for (int ni = 0; ni < 2; ++ni) {
    int n = n0 + ni * 16 + nl;
    float bv = bias ? bias[n] : 0.f;
#pragma unroll
    for (int mi = 0; mi < 2; ++mi) {
      v8f acc = (mi == 0) ? (ni == 0 ? acc00 : acc01) : (ni == 0 ? acc10 : acc11);
      int mb = m0 + mi * 16 + 8 * g;
#pragma unroll
      for (int r = 0; r < 8; ++r) {
        size_t o = (size_t)(mb + r) * N + n;
        float v = acc[r] + bv;
        if (EPI == 0) outF[o] = v;
        else if (EPI == 1) outF[o] = v + res[o];
        else { v = v > 0.f ? v : 0.f; outB[o] = (__bf16)v; }
      }
    }
  }
}

// ---------------- fused scores + bias + softmax + position bins ----------------
// One block: 16 query rows x full 512 keys for one (b,h).
__global__ __launch_bounds__(256) void stsa_scores_softmax(
    const __bf16* q, const __bf16* kx, const float* qz, const float* dist,
    const int* dtime, const int* mask, __bf16* awb, __bf16* sbinb) {
  __shared__ float buf[16][S_];    // 32 KB score stripe
  __shared__ float sbin[16][RP_];  // 10 KB position-bin accumulators
  __shared__ int dts[S_];
  __shared__ int msk[S_];
  __shared__ float red[16][16];
  __shared__ float rowv[16];

  int tid = threadIdx.x;
  int blk = blockIdx.x;
  int bh = blk >> 5;             // S_/16 == 32 tiles per (b,h)
  int i0 = (blk & 31) << 4;
  int b = bh >> 3;               // H_ == 8

  for (int t = tid; t < S_; t += 256) {
    dts[t] = dtime[b * S_ + t];
    msk[t] = mask[b * S_ + t];
  }
  for (int t = tid; t < 16 * RP_; t += 256) (&sbin[0][0])[t] = 0.f;
  __syncthreads();

  int lane = tid & 31;
  int wave = tid >> 5;
  int g = (lane >> 4) & 1;
  int nn = lane & 15;
  const __bf16* Ab = q + ((size_t)bh * S_ + i0) * D_;
  const __bf16* Bbase = kx + (size_t)bh * S_ * D_;

  // load the q fragment once (K = D = 64 -> two 16x32 fragments), reuse for 4 tiles
  v16bf aK0 = load_a_frag(Ab, D_, lane, 0);
  v16bf aK1 = load_a_frag(Ab, D_, lane, 32);

#pragma unroll
  for (int t = 0; t < 4; ++t) {
    int n0 = wave * 64 + t * 16;
    const __bf16* Bb = Bbase + (size_t)n0 * D_;
    v8f acc = {};
    acc = wmma_bf16(aK0, load_b_frag(Bb, D_, lane, 0), acc);
    acc = wmma_bf16(aK1, load_b_frag(Bb, D_, lane, 32), acc);
    int j = n0 + nn;
#pragma unroll
    for (int r = 0; r < 8; ++r) {
      int m = r + 8 * g;
      int i = i0 + m;
      int rel = dts[j] - dts[i];
      rel = rel < -P_ ? -P_ : (rel > P_ ? P_ : rel);
      float val = acc[r] * 0.125f                                  // D^-0.5
                + qz[((size_t)bh * S_ + i) * RP_ + (rel + P_)]     // q . z bias
                + 0.6f * dist[((size_t)b * S_ + i) * S_ + j];
      if (msk[j] == 0) val = -1.0e9f;
      buf[m][j] = val;
    }
  }
  __syncthreads();

  // softmax: 16 threads per row, 32 elements each
  int row = tid >> 4;
  int sub = tid & 15;
  float lmax = -3.4e38f;
#pragma unroll
  for (int e = 0; e < 32; ++e) lmax = fmaxf(lmax, buf[row][sub + 16 * e]);
  red[row][sub] = lmax;
  __syncthreads();
  if (sub == 0) {
    float m2 = red[row][0];
#pragma unroll
    for (int u = 1; u < 16; ++u) m2 = fmaxf(m2, red[row][u]);
    rowv[row] = m2;
  }
  __syncthreads();
  float rmax = rowv[row];
  float lsum = 0.f;
#pragma unroll
  for (int e = 0; e < 32; ++e) {
    int j = sub + 16 * e;
    float ex = __expf(buf[row][j] - rmax);
    buf[row][j] = ex;
    lsum += ex;
  }
  __syncthreads();
  red[row][sub] = lsum;
  __syncthreads();
  if (sub == 0) {
    float s2 = 0.f;
#pragma unroll
    for (int u = 0; u < 16; ++u) s2 += red[row][u];
    rowv[row] = 1.f / s2;
  }
  __syncthreads();
  float inv = rowv[row];
  int i = i0 + row;
  int dti = dts[i];
  for (int e = 0; e < 32; ++e) {
    int j = sub + 16 * e;
    float p = buf[row][j] * inv;
    awb[((size_t)bh * S_ + i) * S_ + j] = (__bf16)p;
    int rel = dts[j] - dti;
    rel = rel < -P_ ? -P_ : (rel > P_ ? P_ : rel);
    atomicAdd(&sbin[row][rel + P_], p);   // ds_add_f32
  }
  __syncthreads();
  for (int t = tid; t < 16 * RP_; t += 256) {
    int rr = t / RP_, cc = t - rr * RP_;
    sbinb[((size_t)bh * S_ + i0 + rr) * RP_ + cc] = (__bf16)sbin[rr][cc];
  }
}

// ---------------- attention output: aw @ v + bins @ pos -> concat ----------------
// 2 waves per block; each wave owns a 16x32 slab (A fragment reused for 2 n-tiles).
__global__ __launch_bounds__(64) void stsa_attn_out(const __bf16* awb, const __bf16* vt,
                                                    const __bf16* sbinb,
                                                    const __bf16* posT, __bf16* concat) {
  int blk = blockIdx.x;
  int bh = blk >> 5;
  int i0 = (blk & 31) << 4;
  int wave = threadIdx.x >> 5;   // 0..1
  int lane = threadIdx.x & 31;
  int n0 = wave * 32;            // d-slab

  const __bf16* A1 = awb + ((size_t)bh * S_ + i0) * S_;
  const __bf16* B1a = vt + ((size_t)bh * D_ + n0) * S_;
  const __bf16* B1b = B1a + (size_t)16 * S_;
  v8f acc0 = {}, acc1 = {};
  int ml = lane & 15;
  for (int k0 = 0; k0 < S_; k0 += 32) {
    if (k0 + 128 < S_)
      __builtin_prefetch(A1 + (size_t)ml * S_ + k0 + 128, 0, 3);
    v16bf a = load_a_frag(A1, S_, lane, k0);
    acc0 = wmma_bf16(a, load_b_frag(B1a, S_, lane, k0), acc0);
    acc1 = wmma_bf16(a, load_b_frag(B1b, S_, lane, k0), acc1);
  }
  const __bf16* A2 = sbinb + ((size_t)bh * S_ + i0) * RP_;
  const __bf16* B2a = posT + (size_t)n0 * RP_;
  const __bf16* B2b = B2a + (size_t)16 * RP_;
#pragma unroll
  for (int k0 = 0; k0 < RP_; k0 += 32) {
    v16bf a = load_a_frag(A2, RP_, lane, k0);
    acc0 = wmma_bf16(a, load_b_frag(B2a, RP_, lane, k0), acc0);
    acc1 = wmma_bf16(a, load_b_frag(B2b, RP_, lane, k0), acc1);
  }
  int g = (lane >> 4) & 1;
  int b0 = bh >> 3, h = bh & 7;
#pragma unroll
  for (int ni = 0; ni < 2; ++ni) {
    int d = n0 + ni * 16 + (lane & 15);
    v8f acc = ni ? acc1 : acc0;
#pragma unroll
    for (int r = 0; r < 8; ++r) {
      int i = i0 + r + 8 * g;
      concat[((size_t)b0 * S_ + i) * E_ + h * D_ + d] = (__bf16)acc[r];
    }
  }
}

// ---------------- layernorm over E=512 ----------------
__global__ __launch_bounds__(256) void stsa_layernorm(const float* y, const float* gg,
                                                      const float* bb, float* outF,
                                                      __bf16* outB) {
  __shared__ float red[256];
  int row = blockIdx.x, tid = threadIdx.x;
  const float* yr = y + (size_t)row * E_;
  float a0 = yr[tid], a1 = yr[tid + 256];
  red[tid] = a0 + a1;
  __syncthreads();
  for (int s = 128; s > 0; s >>= 1) {
    if (tid < s) red[tid] += red[tid + s];
    __syncthreads();
  }
  float mu = red[0] * (1.f / E_);
  __syncthreads();
  float d0 = a0 - mu, d1 = a1 - mu;
  red[tid] = d0 * d0 + d1 * d1;
  __syncthreads();
  for (int s = 128; s > 0; s >>= 1) {
    if (tid < s) red[tid] += red[tid + s];
    __syncthreads();
  }
  float inv = rsqrtf(red[0] * (1.f / E_) + 1e-5f);
  float o0 = d0 * inv * gg[tid] + bb[tid];
  float o1 = d1 * inv * gg[tid + 256] + bb[tid + 256];
  size_t o = (size_t)row * E_ + tid;
  outF[o] = o0;
  outF[o + 256] = o1;
  if (outB) {
    outB[o] = (__bf16)o0;
    outB[o + 256] = (__bf16)o1;
  }
}

// ---------------- host launcher ----------------
extern "C" void kernel_launch(void* const* d_in, const int* in_sizes, int n_in,
                              void* d_out, int out_size, void* d_ws, size_t ws_size,
                              hipStream_t stream) {
  const float* x     = (const float*)d_in[0];
  const float* dist  = (const float*)d_in[1];
  const int*   dtm   = (const int*)d_in[2];
  const int*   mask  = (const int*)d_in[3];
  const float* wq_w  = (const float*)d_in[4];
  const float* wq_b  = (const float*)d_in[5];
  const float* wk_w  = (const float*)d_in[6];
  const float* wk_b  = (const float*)d_in[7];
  const float* wv_w  = (const float*)d_in[8];
  const float* wv_b  = (const float*)d_in[9];
  const float* wo_w  = (const float*)d_in[10];
  const float* wo_b  = (const float*)d_in[11];
  const float* pos   = (const float*)d_in[12];
  const float* w1    = (const float*)d_in[13];
  const float* b1    = (const float*)d_in[14];
  const float* w2    = (const float*)d_in[15];
  const float* b2    = (const float*)d_in[16];
  const float* ln1g  = (const float*)d_in[17];
  const float* ln1b  = (const float*)d_in[18];
  const float* ln2g  = (const float*)d_in[19];
  const float* ln2b  = (const float*)d_in[20];
  float* out = (float*)d_out;

  char* ws = (char*)d_ws;
  size_t off = 0;
  auto take = [&](size_t bytes) -> char* {
    char* p = ws + off;
    off = (off + bytes + 255) & ~(size_t)255;
    return p;
  };
  __bf16* XB   = (__bf16*)take((size_t)BS_ * E_ * 2);
  __bf16* WQT  = (__bf16*)take((size_t)E_ * E_ * 2);
  __bf16* WKT  = (__bf16*)take((size_t)E_ * E_ * 2);
  __bf16* WVT  = (__bf16*)take((size_t)E_ * E_ * 2);
  __bf16* WOT  = (__bf16*)take((size_t)E_ * E_ * 2);
  __bf16* W1T  = (__bf16*)take((size_t)F_ * E_ * 2);
  __bf16* W2T  = (__bf16*)take((size_t)E_ * F_ * 2);
  __bf16* POSB = (__bf16*)take((size_t)RP_ * D_ * 2);
  __bf16* POST = (__bf16*)take((size_t)D_ * RP_ * 2);
  __bf16* Q    = (__bf16*)take((size_t)BH_ * S_ * D_ * 2);
  __bf16* KX   = (__bf16*)take((size_t)BH_ * S_ * D_ * 2);
  __bf16* VT   = (__bf16*)take((size_t)BH_ * D_ * S_ * 2);
  float*  QZ   = (float*)take((size_t)BH_ * S_ * RP_ * 4);
  __bf16* AWB  = (__bf16*)take((size_t)BH_ * S_ * S_ * 2);
  __bf16* SBB  = (__bf16*)take((size_t)BH_ * S_ * RP_ * 2);
  __bf16* CONC = (__bf16*)take((size_t)BS_ * E_ * 2);
  float*  Y1   = (float*)take((size_t)BS_ * E_ * 4);
  float*  LN1F = (float*)take((size_t)BS_ * E_ * 4);
  __bf16* LN1B = (__bf16*)take((size_t)BS_ * E_ * 2);
  __bf16* HB   = (__bf16*)take((size_t)BS_ * F_ * 2);
  float*  Y2   = (float*)take((size_t)BS_ * E_ * 4);

  // 1) conversions
  stsa_cvt_bf16<<<(BS_ * E_ + 255) / 256, 256, 0, stream>>>(x, XB, BS_ * E_);
  stsa_transpose_bf16<<<(E_ * E_ + 255) / 256, 256, 0, stream>>>(wq_w, WQT, E_, E_);
  stsa_transpose_bf16<<<(E_ * E_ + 255) / 256, 256, 0, stream>>>(wk_w, WKT, E_, E_);
  stsa_transpose_bf16<<<(E_ * E_ + 255) / 256, 256, 0, stream>>>(wv_w, WVT, E_, E_);
  stsa_transpose_bf16<<<(E_ * E_ + 255) / 256, 256, 0, stream>>>(wo_w, WOT, E_, E_);
  stsa_transpose_bf16<<<(E_ * F_ + 255) / 256, 256, 0, stream>>>(w1, W1T, E_, F_);
  stsa_transpose_bf16<<<(F_ * E_ + 255) / 256, 256, 0, stream>>>(w2, W2T, F_, E_);
  stsa_pos_cvt<<<(RP_ * D_ + 255) / 256, 256, 0, stream>>>(pos, POSB, POST);

  // 2) QKV projections (head-major; V transposed for the PV GEMM)
  dim3 gQKV(BS_ / 32, E_ / 128);
  stsa_gemm_qkv<<<gQKV, 128, 0, stream>>>(XB, WQT, wq_b, Q, 0);
  stsa_gemm_qkv<<<gQKV, 128, 0, stream>>>(XB, WKT, wk_b, KX, 0);
  stsa_gemm_qkv<<<gQKV, 128, 0, stream>>>(XB, WVT, wv_b, VT, 1);

  // 3) qz[bh,i,r] = q . pos_emb^T  (M=BH*S rows, N=RP, K=D)
  dim3 gQZ((BH_ * S_) / 32, (RP_ + 127) / 128);
  stsa_gemm<0><<<gQZ, 128, 0, stream>>>(Q, POSB, nullptr, nullptr, QZ, nullptr,
                                        RP_, D_, D_, D_);

  // 4) fused scores + softmax + position bins
  stsa_scores_softmax<<<BH_ * (S_ / 16), 256, 0, stream>>>(Q, KX, QZ, dist, dtm, mask,
                                                           AWB, SBB);

  // 5) attn = aw @ v + bins @ pos  -> concat (B,S,E) bf16
  stsa_attn_out<<<BH_ * (S_ / 16), 64, 0, stream>>>(AWB, VT, SBB, POST, CONC);

  // 6) O-projection + residual -> Y1
  dim3 gO(BS_ / 32, E_ / 128);
  stsa_gemm<1><<<gO, 128, 0, stream>>>(CONC, WOT, wo_b, x, Y1, nullptr,
                                       E_, E_, E_, E_);
  // 7) LN1
  stsa_layernorm<<<BS_, 256, 0, stream>>>(Y1, ln1g, ln1b, LN1F, LN1B);

  // 8) FFN1 (relu, bf16 out)
  dim3 gF1(BS_ / 32, F_ / 128);
  stsa_gemm<2><<<gF1, 128, 0, stream>>>(LN1B, W1T, b1, nullptr, nullptr, HB,
                                        F_, E_, E_, E_);
  // 9) FFN2 + residual
  dim3 gF2(BS_ / 32, E_ / 128);
  stsa_gemm<1><<<gF2, 128, 0, stream>>>(HB, W2T, b2, LN1F, Y2, nullptr,
                                        E_, F_, F_, F_);
  // 10) LN2 -> output
  stsa_layernorm<<<BS_, 256, 0, stream>>>(Y2, ln2g, ln2b, out, nullptr);
}
